// Generator_63831803953476
// MI455X (gfx1250) — compile-verified, matrix operand order checked by reading, not run on hardware
//
#include <hip/hip_runtime.h>
#include <hip/hip_bf16.h>
#include <hip/hip_fp16.h>
#include <math.h>

// ---------------------------------------------------------------------------
// LSTM language model forward for MI455X (gfx1250), wave32 + WMMA f16->f32.
// B=16, S=256, V=32000, E=512, H=1024, 2 layers.
// Big GEMMs are register-blocked: 32x64 output per wave (2x4 WMMA tiles),
// ~22 flops/byte; compiler schedules load/WMMA overlap (manual double-
// buffering produced mov/nop churn on this toolchain, so we let it).
// ---------------------------------------------------------------------------

typedef __attribute__((ext_vector_type(16))) _Float16 v16h;
typedef __attribute__((ext_vector_type(8)))  _Float16 v8h;
typedef __attribute__((ext_vector_type(8)))  float    v8f;

static constexpr int B_  = 16;
static constexpr int S_  = 256;
static constexpr int V_  = 32000;
static constexpr int E_  = 512;
static constexpr int H_  = 1024;
static constexpr int G4H = 4 * H_;   // 4096
static constexpr int BS  = B_ * S_;  // 4096

// ---------------------------------------------------------------------------
// A/B tile loader for V_WMMA_F32_16X16X32_F16.
// 16-bit A-matrix 16x32 layout (ISA 7.12.2): lanes 0-15 -> M=lane,
//   VGPR0..3 = K 0..7, VGPR4..7 = K 16..23; lanes 16-31 -> M=lane-16,
//   VGPR0..3 = K 8..15, VGPR4..7 = K 24..31.
// => per lane: two contiguous 8-half (16B) runs at k0+8*kh and k0+16+8*kh.
// B is loaded identically from the transposed weight [N][K].
// ---------------------------------------------------------------------------
__device__ __forceinline__ v16h load_tile_f16(const _Float16* __restrict__ base,
                                              int ld, int row0, int k0, int lane) {
  const int r  = lane & 15;
  const int kh = lane >> 4;
  const _Float16* p = base + (size_t)(row0 + r) * ld + k0 + 8 * kh;
  v8h lo = *(const v8h*)(p);
  v8h hi = *(const v8h*)(p + 16);
  v16h a;
#pragma unroll
  for (int i = 0; i < 8; ++i) { a[i] = lo[i]; a[i + 8] = hi[i]; }
  return a;
}

// ---------------------------------------------------------------------------
// Register-blocked GEMM: C[M,N] = A[M,K](f16) * B[K,N] + bias, with B given
// transposed as BT[N,K] f16. Each wave computes a 32x64 output block as
// 2(M) x 4(N) WMMA tiles; A tiles reused 4x, B tiles reused 2x per K-step.
// Requires M%32==0, N%64==0, K%32==0.
// ---------------------------------------------------------------------------
__global__ void gemm_wmma_f16_blk(const _Float16* __restrict__ A,
                                  const _Float16* __restrict__ BT,
                                  const float* __restrict__ bias,
                                  float* __restrict__ C,
                                  int M, int N, int K) {
  const int wv   = threadIdx.x >> 5;
  const int lane = threadIdx.x & 31;
  const int bn_tiles = N >> 6;                       // 64-wide blocks
  const long total   = (long)(M >> 5) * bn_tiles;    // 32-tall blocks
  const long blk     = (long)blockIdx.x * 4 + wv;
  if (blk >= total) return;                          // wave-uniform: EXEC all-1s
  const int m0 = (int)(blk / bn_tiles) << 5;
  const int n0 = (int)(blk % bn_tiles) << 6;

  v8f acc[2][4] = {};
  for (int k0 = 0; k0 < K; k0 += 32) {
    v16h a0 = load_tile_f16(A, K, m0,      k0, lane);
    v16h a1 = load_tile_f16(A, K, m0 + 16, k0, lane);
    v16h b[4];
#pragma unroll
    for (int j = 0; j < 4; ++j)
      b[j] = load_tile_f16(BT, K, n0 + 16 * j, k0, lane);
#pragma unroll
    for (int j = 0; j < 4; ++j) {
      acc[0][j] = __builtin_amdgcn_wmma_f32_16x16x32_f16(false, a0, false, b[j],
                                                         (short)0, acc[0][j], false, false);
      acc[1][j] = __builtin_amdgcn_wmma_f32_16x16x32_f16(false, a1, false, b[j],
                                                         (short)0, acc[1][j], false, false);
    }
  }
  // C/D layout: VGPR r -> M = mtile + r + 8*(lane>>4), N = ntile + (lane&15)
  const int cc = lane & 15;
  const int kh = lane >> 4;
#pragma unroll
  for (int i = 0; i < 2; ++i) {
#pragma unroll
    for (int j = 0; j < 4; ++j) {
      const int col = n0 + 16 * j + cc;
      const float bv = bias ? bias[col] : 0.0f;
#pragma unroll
      for (int r = 0; r < 8; ++r)
        C[(size_t)(m0 + 16 * i + r + 8 * kh) * N + col] = acc[i][j][r] + bv;
    }
  }
}

// ---------------------------------------------------------------------------
// One LSTM timestep. Grid: H/16 = 64 blocks, 128 threads = 4 waves.
// Wave g (g = i,f,g,o) computes the 16(batch) x 16(hidden) z-tile at column
// g*H + j0 of  z = xW[:,t,:] + h @ U  (xW already contains x@W + b).
// Gates exchanged via LDS, then fused elementwise LSTM cell update with mask.
// h for the recurrent matmul is double-buffered in f16 (hin/hout).
// ---------------------------------------------------------------------------
__global__ void lstm_step(const float* __restrict__ xW,        // [BS][4H]
                          const _Float16* __restrict__ UT,     // [4H][H]
                          const _Float16* __restrict__ hin,    // [B][H] f16
                          _Float16* __restrict__ hout,         // [B][H] f16
                          float* __restrict__ h_f32,           // [B][H]
                          float* __restrict__ c_f32,           // [B][H]
                          _Float16* __restrict__ ys,           // [BS][H] f16
                          const int* __restrict__ tokens,      // [B][S]
                          int t) {
  __shared__ float zsh[4][16][16];
  const int wv   = threadIdx.x >> 5;
  const int lane = threadIdx.x & 31;
  const int j0   = blockIdx.x << 4;
  const int n0   = wv * H_ + j0;

  v8f acc = {};
  for (int k0 = 0; k0 < H_; k0 += 32) {
    v16h a = load_tile_f16(hin, H_, 0,  k0, lane);
    v16h b = load_tile_f16(UT,  H_, n0, k0, lane);
    acc = __builtin_amdgcn_wmma_f32_16x16x32_f16(false, a, false, b,
                                                 (short)0, acc, false, false);
  }
  const int cc = lane & 15;
  const int kh = lane >> 4;
#pragma unroll
  for (int r = 0; r < 8; ++r) {
    const int m = r + 8 * kh;                       // batch row
    zsh[wv][m][cc] = acc[r] + xW[(size_t)(m * S_ + t) * G4H + n0 + cc];
  }
  __syncthreads();

  for (int e = threadIdx.x; e < 256; e += blockDim.x) {
    const int m  = e >> 4;
    const int jj = e & 15;
    const int j  = j0 + jj;
    const float zi = zsh[0][m][jj];
    const float zf = zsh[1][m][jj];
    const float zg = zsh[2][m][jj];
    const float zo = zsh[3][m][jj];
    const float co = c_f32[m * H_ + j];
    const float ho = h_f32[m * H_ + j];
    const float ig = 1.0f / (1.0f + __expf(-zi));
    const float fg = 1.0f / (1.0f + __expf(-zf));
    const float gg = tanhf(zg);
    const float og = 1.0f / (1.0f + __expf(-zo));
    float cn = fg * co + ig * gg;
    float hn = og * tanhf(cn);
    const bool msk = tokens[m * S_ + t] != 0;
    hn = msk ? hn : ho;
    cn = msk ? cn : co;
    c_f32[m * H_ + j] = cn;
    h_f32[m * H_ + j] = hn;
    hout[m * H_ + j]  = (_Float16)hn;
    ys[(size_t)(m * S_ + t) * H_ + j] = (_Float16)hn;
  }
}

// ---------------------------------------------------------------------------
// Embedding gather + cast to f16: x0[row][e] = emb[tokens[row]][e]
// ---------------------------------------------------------------------------
__global__ void gather_emb(const int* __restrict__ tokens,
                           const float* __restrict__ emb,
                           _Float16* __restrict__ x0, long total, int ecols) {
  long i = (long)blockIdx.x * blockDim.x + threadIdx.x;
  if (i >= total) return;
  const int row = (int)(i / ecols);
  const int e   = (int)(i % ecols);
  x0[i] = (_Float16)emb[(size_t)tokens[row] * ecols + e];
}

// ---------------------------------------------------------------------------
// Transpose + cast: in[K][N] f32 -> out[N][K] f16
// ---------------------------------------------------------------------------
__global__ void transpose_cast(const float* __restrict__ in,
                               _Float16* __restrict__ out, long total,
                               int K, int N) {
  long i = (long)blockIdx.x * blockDim.x + threadIdx.x;
  if (i >= total) return;
  const int k = (int)(i / N);
  const int n = (int)(i % N);
  out[(size_t)n * K + k] = (_Float16)in[i];
}

// ---------------------------------------------------------------------------
// In-place row softmax, one 256-thread block per row of length N.
// ---------------------------------------------------------------------------
__global__ void softmax_rows(float* __restrict__ x, int N) {
  float* p = x + (size_t)blockIdx.x * N;
  __shared__ float red[256];
  float mx = -3.402823466e38f;
  for (int i = threadIdx.x; i < N; i += blockDim.x) mx = fmaxf(mx, p[i]);
  red[threadIdx.x] = mx; __syncthreads();
  for (int s = blockDim.x >> 1; s; s >>= 1) {
    if (threadIdx.x < s) red[threadIdx.x] = fmaxf(red[threadIdx.x], red[threadIdx.x + s]);
    __syncthreads();
  }
  mx = red[0]; __syncthreads();
  float sum = 0.0f;
  for (int i = threadIdx.x; i < N; i += blockDim.x) {
    const float e = __expf(p[i] - mx);
    p[i] = e;
    sum += e;
  }
  red[threadIdx.x] = sum; __syncthreads();
  for (int s = blockDim.x >> 1; s; s >>= 1) {
    if (threadIdx.x < s) red[threadIdx.x] += red[threadIdx.x + s];
    __syncthreads();
  }
  const float inv = 1.0f / red[0];
  for (int i = threadIdx.x; i < N; i += blockDim.x) p[i] *= inv;
}

// ---------------------------------------------------------------------------
// Pack final h/c stacks: out = [h0, h1, c0, c1], each B*H floats.
// ---------------------------------------------------------------------------
__global__ void copy_states(const float* __restrict__ h0, const float* __restrict__ h1,
                            const float* __restrict__ c0, const float* __restrict__ c1,
                            float* __restrict__ out) {
  const int n = B_ * H_;
  const int i = blockIdx.x * blockDim.x + threadIdx.x;
  if (i < n) {
    out[i]         = h0[i];
    out[n + i]     = h1[i];
    out[2 * n + i] = c0[i];
    out[3 * n + i] = c1[i];
  }
}

// ---------------------------------------------------------------------------
// Host-side orchestration (graph-capture safe: only async ops on `stream`).
// ---------------------------------------------------------------------------
extern "C" void kernel_launch(void* const* d_in, const int* in_sizes, int n_in,
                              void* d_out, int out_size, void* d_ws, size_t ws_size,
                              hipStream_t stream) {
  (void)in_sizes; (void)n_in; (void)out_size; (void)ws_size;
  const int*   tokens = (const int*)  d_in[0];   // [B,S]
  const float* emb    = (const float*)d_in[1];   // [V,E]
  const float* W0     = (const float*)d_in[2];   // [E,4H]
  const float* U0     = (const float*)d_in[3];   // [H,4H]
  const float* b0     = (const float*)d_in[4];   // [4H]
  const float* W1     = (const float*)d_in[5];   // [H,4H]
  const float* U1     = (const float*)d_in[6];   // [H,4H]
  const float* b1     = (const float*)d_in[7];   // [4H]
  const float* Wd     = (const float*)d_in[8];   // [H,V]
  const float* bd     = (const float*)d_in[9];   // [V]

  // ---- workspace carve-out (all offsets 256B aligned) ----
  char* ws = (char*)d_ws;
  size_t off = 0;
  auto carve = [&](size_t bytes) -> char* {
    char* p = ws + off;
    off = (off + bytes + 255) & ~(size_t)255;
    return p;
  };
  _Float16* W0T  = (_Float16*)carve((size_t)G4H * E_ * 2);   // [4H][E]
  _Float16* U0T  = (_Float16*)carve((size_t)G4H * H_ * 2);   // [4H][H]
  _Float16* W1T  = (_Float16*)carve((size_t)G4H * H_ * 2);   // [4H][H]
  _Float16* U1T  = (_Float16*)carve((size_t)G4H * H_ * 2);   // [4H][H]
  _Float16* WdT  = (_Float16*)carve((size_t)V_  * H_ * 2);   // [V][H]
  _Float16* x0   = (_Float16*)carve((size_t)BS  * E_ * 2);   // [BS][E]
  float*    xW   = (float*)   carve((size_t)BS  * G4H * 4);  // [BS][4H] (reused L0/L1)
  _Float16* ys0  = (_Float16*)carve((size_t)BS  * H_ * 2);   // [BS][H]
  _Float16* ys1  = (_Float16*)carve((size_t)BS  * H_ * 2);   // [BS][H]
  float*    h0s  = (float*)   carve((size_t)B_ * H_ * 4);
  float*    c0s  = (float*)   carve((size_t)B_ * H_ * 4);
  float*    h1s  = (float*)   carve((size_t)B_ * H_ * 4);
  float*    c1s  = (float*)   carve((size_t)B_ * H_ * 4);
  _Float16* hA   = (_Float16*)carve((size_t)B_ * H_ * 2);    // recurrent ping
  _Float16* hB   = (_Float16*)carve((size_t)B_ * H_ * 2);    // recurrent pong

  float* probs  = (float*)d_out;                       // [BS][V]
  float* states = (float*)d_out + (size_t)BS * V_;     // [4][B*H] = h0,h1,c0,c1

  const int TB = 256;
  auto blocks = [](long total, int tb) { return (unsigned)((total + tb - 1) / tb); };

  // ---- 1. weight transposes (f32 [K][N] -> f16 [N][K]) ----
  {
    long n;
    n = (long)E_ * G4H; transpose_cast<<<blocks(n, TB), TB, 0, stream>>>(W0, W0T, n, E_, G4H);
    n = (long)H_ * G4H; transpose_cast<<<blocks(n, TB), TB, 0, stream>>>(U0, U0T, n, H_, G4H);
    n = (long)H_ * G4H; transpose_cast<<<blocks(n, TB), TB, 0, stream>>>(W1, W1T, n, H_, G4H);
    n = (long)H_ * G4H; transpose_cast<<<blocks(n, TB), TB, 0, stream>>>(U1, U1T, n, H_, G4H);
    n = (long)H_ * V_;  transpose_cast<<<blocks(n, TB), TB, 0, stream>>>(Wd, WdT, n, H_, V_);
  }

  // ---- 2. embedding gather ----
  {
    long n = (long)BS * E_;
    gather_emb<<<blocks(n, TB), TB, 0, stream>>>(tokens, emb, x0, n, E_);
  }

  // ---- 3. xW0 = x0 @ W0 + b0 ----
  {
    long wvs = (long)(BS / 32) * (G4H / 64);
    gemm_wmma_f16_blk<<<blocks(wvs, 4), 128, 0, stream>>>(x0, W0T, b0, xW, BS, G4H, E_);
  }

  // ---- 4. layer-0 recurrence ----
  hipMemsetAsync(h0s, 0, (size_t)B_ * H_ * 4, stream);
  hipMemsetAsync(c0s, 0, (size_t)B_ * H_ * 4, stream);
  hipMemsetAsync(hA,  0, (size_t)B_ * H_ * 2, stream);
  for (int t = 0; t < S_; ++t) {
    _Float16* hin  = (t & 1) ? hB : hA;
    _Float16* hout = (t & 1) ? hA : hB;
    lstm_step<<<H_ / 16, 128, 0, stream>>>(xW, U0T, hin, hout, h0s, c0s, ys0, tokens, t);
  }

  // ---- 5. xW1 = ys0 @ W1 + b1 (reuse xW buffer) ----
  {
    long wvs = (long)(BS / 32) * (G4H / 64);
    gemm_wmma_f16_blk<<<blocks(wvs, 4), 128, 0, stream>>>(ys0, W1T, b1, xW, BS, G4H, H_);
  }

  // ---- 6. layer-1 recurrence ----
  hipMemsetAsync(h1s, 0, (size_t)B_ * H_ * 4, stream);
  hipMemsetAsync(c1s, 0, (size_t)B_ * H_ * 4, stream);
  hipMemsetAsync(hA,  0, (size_t)B_ * H_ * 2, stream);
  for (int t = 0; t < S_; ++t) {
    _Float16* hin  = (t & 1) ? hB : hA;
    _Float16* hout = (t & 1) ? hA : hB;
    lstm_step<<<H_ / 16, 128, 0, stream>>>(xW, U1T, hin, hout, h1s, c1s, ys1, tokens, t);
  }

  // ---- 7. logits = ys1 @ Wd + bd (straight into d_out) ----
  {
    long wvs = (long)(BS / 32) * (V_ / 64);
    gemm_wmma_f16_blk<<<blocks(wvs, 4), 128, 0, stream>>>(ys1, WdT, bd, probs, BS, V_, H_);
  }

  // ---- 8. softmax in place ----
  softmax_rows<<<BS, 256, 0, stream>>>(probs, V_);

  // ---- 9. final h/c stacks ----
  copy_states<<<blocks(B_ * H_, TB), TB, 0, stream>>>(h0s, h1s, c0s, c1s, states);
}
